// SP_4D_8057358647955
// MI455X (gfx1250) — matrix operand order repaired and checked by reading, not compiled, not run on hardware
//
#include <hip/hip_runtime.h>
#include <hip/hip_bf16.h>

typedef __attribute__((ext_vector_type(16))) _Float16 v16h;
typedef __attribute__((ext_vector_type(8)))  _Float16 v8h;
typedef __attribute__((ext_vector_type(8)))  float    v8f;

// ---------------- geometry ----------------
#define NB      128
#define CH      16
#define NPOS    8100            // 10*10*9*9 output positions per batch
#define T32     254             // ceil(8100/32) 32-position tiles (2 WMMA M-tiles each)
// padded f16 activation tensor: [B][12][12][11][11][16] (zero halo of 1)
#define STR_H   16
#define STR_G   176             // 11*16
#define STR_F   1936            // 11*11*16
#define STR_E   23232           // 12*11*11*16
#define STR_B   278784          // 12*12*11*11*16
#define ACT_HALVES (NB * STR_B)           // 35,684,352 halves per buffer
#define NTAPS     81
#define NTAPS_PAD 82
#define WL_HALVES (NTAPS_PAD * CH * CH)   // 20,992 halves per layer

// ---------------- zero fill (halo init, deterministic each call) ----------------
__global__ void zero_ws_kernel(float4* __restrict__ p, int n) {
    int i = blockIdx.x * blockDim.x + threadIdx.x;
    int stride = gridDim.x * blockDim.x;
    float4 z = make_float4(0.f, 0.f, 0.f, 0.f);
    for (; i < n; i += stride) p[i] = z;
}

// ---------------- weight transpose + f16: wf[layer][tap][co][ci] ----------------
__global__ void prep_weights_kernel(const float* __restrict__ w1,
                                    const float* __restrict__ w2,
                                    const float* __restrict__ w3,
                                    const float* __restrict__ w4,
                                    const float* __restrict__ w5,
                                    _Float16* __restrict__ wf) {
    int idx = blockIdx.x * blockDim.x + threadIdx.x;
    const int total = 5 * WL_HALVES;
    if (idx >= total) return;
    int ci = idx & 15;
    int co = (idx >> 4) & 15;
    int t  = (idx >> 8) % NTAPS_PAD;
    int l  = idx / WL_HALVES;
    float v = 0.f;
    if (t < NTAPS) {
        if (l == 0) {
            if (ci == 0) v = w1[co * NTAPS + t];           // layer1: Cin=1, pad ci 1..15 with 0
        } else {
            const float* w = (l == 1) ? w2 : (l == 2) ? w3 : (l == 3) ? w4 : w5;
            v = w[co * (16 * NTAPS) + ci * NTAPS + t];
        }
    }
    wf[idx] = (_Float16)v;
}

// ---------------- center-subtract prep into padded f16 tensor (channel 0) ----------------
__global__ void prep_input_kernel(const float* __restrict__ x, _Float16* __restrict__ act) {
    int idx = blockIdx.x * blockDim.x + threadIdx.x;     // over 128*100*81
    if (idx >= NB * 100 * 81) return;
    int h    = idx % 9;
    int g    = (idx / 9) % 9;
    int band = (idx / 81) % 100;
    int b    = idx / 8100;
    int e = band / 10, f = band % 10;
    const float* xb = x + (size_t)(b * 100 + band) * 81;
    float v = xb[g * 9 + h] - xb[4 * 9 + 4];
    act[(size_t)b * STR_B + (e + 1) * STR_E + (f + 1) * STR_F +
        (g + 1) * STR_G + (h + 1) * STR_H + 0] = (_Float16)v;
}

// ---------------- tap -> padded-tensor offset (folds to immediates under unroll) ----------------
__device__ __forceinline__ constexpr int tap_off(int t) {
    return (t / 27) * STR_E + ((t / 9) % 3) * STR_F + ((t / 3) % 3) * STR_G + (t % 3) * STR_H;
}

// A-fragment base pointer for a 16-position M-tile starting at pstart
__device__ __forceinline__ const _Float16* frag_base(const _Float16* __restrict__ in,
                                                     int b, int pstart, int lane, int halfSel) {
    int pA = pstart + (lane & 15);
    if (pA > NPOS - 1) pA = NPOS - 1;                     // clamp tail tile (loads only)
    int h = pA % 9;
    int q = pA / 9;
    int g = q % 9;  q /= 9;
    int f = q % 10;
    int e = q / 10;
    return in + (size_t)b * STR_B + e * STR_E + f * STR_F +
           g * STR_G + h * STR_H + halfSel * 8;
}

// bias + ReLU + store a v8f accumulator; incremental (h,g,f,e) walk, no div chains
__device__ __forceinline__ void store_acc(_Float16* __restrict__ out, size_t bbase,
                                          int pstart, int co, v8f acc, float bco) {
    int h = pstart % 9;
    int q = pstart / 9;
    int g = q % 9;  q /= 9;
    int f = q % 10;
    int e = q / 10;
#pragma unroll
    for (int r = 0; r < 8; ++r) {
        if (e < 10) {                                      // p < NPOS
            float v = acc[r] + bco;
            v = v > 0.f ? v : 0.f;
            out[bbase + (e + 1) * STR_E + (f + 1) * STR_F +
                (g + 1) * STR_G + (h + 1) * STR_H + co] = (_Float16)v;
        }
        if (++h == 9) { h = 0; if (++g == 9) { g = 0; if (++f == 10) { f = 0; ++e; } } }
    }
}

// ---------------- generic 4D conv + bias + ReLU via WMMA f32<=f16 ----------------
// grid = (ceil(254/8)=32, 128 batches), block = 256 (8 waves; 32 positions per wave)
__global__ __launch_bounds__(256) void conv4d_wmma_kernel(const _Float16* __restrict__ in,
                                                          _Float16* __restrict__ out,
                                                          const _Float16* __restrict__ wl,
                                                          const float* __restrict__ bias) {
    __shared__ _Float16 lw[WL_HALVES];                    // 41,984 B, resident all kernel
    for (int i = threadIdx.x; i < WL_HALVES / 8; i += 256)
        ((v8h*)lw)[i] = ((const v8h*)wl)[i];
    __syncthreads();

    const int wave = threadIdx.x >> 5;
    const int lane = threadIdx.x & 31;
    const int tile = blockIdx.x * 8 + wave;
    if (tile >= T32) return;

    const int b  = blockIdx.y;
    const int p0 = tile * 32;
    const int halfSel = lane >> 4;                        // 0: lanes 0-15, 1: lanes 16-31

    const _Float16* base0 = frag_base(in, b, p0,      lane, halfSel);
    const _Float16* base1 = frag_base(in, b, p0 + 16, lane, halfSel);
    const _Float16* lwl   = &lw[(lane & 15) * 16 + halfSel * 8]; // B col N=lane%16, ci 8h..8h+7; +pair*512 halves

    const float bco = bias[lane & 15];
    v8f acc0 = {}, acc1 = {};

#pragma unroll
    for (int pair = 0; pair < 41; ++pair) {
        const int t0 = pair * 2;
        const int t1 = (pair * 2 + 1 > NTAPS - 1) ? NTAPS - 1 : pair * 2 + 1; // tap 81 wt == 0
        const int o0 = tap_off(t0);
        const int o1 = tap_off(t1);
        // A fragments: two taps x 8 channels each, immediate offsets off one base VGPR pair
        v16h a0 = __builtin_shufflevector(*(const v8h*)(base0 + o0), *(const v8h*)(base0 + o1),
                                          0, 1, 2, 3, 4, 5, 6, 7, 8, 9, 10, 11, 12, 13, 14, 15);
        v16h a1 = __builtin_shufflevector(*(const v8h*)(base1 + o0), *(const v8h*)(base1 + o1),
                                          0, 1, 2, 3, 4, 5, 6, 7, 8, 9, 10, 11, 12, 13, 14, 15);
        // B fragment: col N = lane%16 of [tap][co][ci] in LDS, taps t0/t0+1, shared by both tiles
        const _Float16* wp = lwl + pair * 512;
        v16h bm = __builtin_shufflevector(*(const v8h*)wp, *(const v8h*)(wp + 256),
                                          0, 1, 2, 3, 4, 5, 6, 7, 8, 9, 10, 11, 12, 13, 14, 15);
        acc0 = __builtin_amdgcn_wmma_f32_16x16x32_f16(false, a0, false, bm,
                                                      (short)0, acc0, false, false);
        acc1 = __builtin_amdgcn_wmma_f32_16x16x32_f16(false, a1, false, bm,
                                                      (short)0, acc1, false, false);
        // Tie the two accumulator chains together every iteration (zero-cost) so IR-level
        // code motion cannot sink the acc1 chain away from the shared B fragment; this
        // bounds each B fragment's live range to one iteration (kills scratch spills).
        asm volatile("" : "+v"(acc0), "+v"(acc1));
        // MIR scheduler: pin WMMA + DS in-iteration, allow ALU and VMEM reads (A stream)
        // to hoist across iterations for latency hiding.
        __builtin_amdgcn_sched_barrier(0x0027);
    }

    const size_t bbase = (size_t)b * STR_B;
    const int co = lane & 15;
    store_acc(out, bbase, p0 + halfSel * 8,      co, acc0, bco);
    store_acc(out, bbase, p0 + 16 + halfSel * 8, co, acc1, bco);
}

// ---------------- mean over (e,f): feat[b][g][h][c] fp32 ----------------
__global__ void mean_ef_kernel(const _Float16* __restrict__ act, float* __restrict__ feat) {
    int idx = blockIdx.x * blockDim.x + threadIdx.x;      // over 128*81*16
    if (idx >= NB * 81 * CH) return;
    int c = idx & 15;
    int h = (idx >> 4) % 9;
    int g = (idx / (16 * 9)) % 9;
    int b = idx / (16 * 81);
    size_t base = (size_t)b * STR_B + (g + 1) * STR_G + (h + 1) * STR_H + c;
    float s = 0.f;
    for (int e = 0; e < 10; ++e)
        for (int f = 0; f < 10; ++f)
            s += (float)act[base + (e + 1) * STR_E + (f + 1) * STR_F];
    feat[idx] = s * 0.01f;
}

// ---------------- FC: out[b][o][g][h] = feat[b][g][h][:] . fc_w[o][:] + fc_b[o] ----------------
__global__ void fc_kernel(const float* __restrict__ feat, const float* __restrict__ fc_w,
                          const float* __restrict__ fc_b, float* __restrict__ out) {
    int idx = blockIdx.x * blockDim.x + threadIdx.x;      // over 128*10*81, matches out layout
    if (idx >= NB * 10 * 81) return;
    int h = idx % 9;
    int g = (idx / 9) % 9;
    int o = (idx / 81) % 10;
    int b = idx / 810;
    const float* fp = feat + ((size_t)(b * 9 + g) * 9 + h) * CH;
    float s = fc_b[o];
#pragma unroll
    for (int c = 0; c < CH; ++c) s += fp[c] * fc_w[o * CH + c];
    out[idx] = s;
}

extern "C" void kernel_launch(void* const* d_in, const int* in_sizes, int n_in,
                              void* d_out, int out_size, void* d_ws, size_t ws_size,
                              hipStream_t stream) {
    const float* x    = (const float*)d_in[0];
    const float* w1   = (const float*)d_in[1];
    const float* b1   = (const float*)d_in[2];
    const float* w2   = (const float*)d_in[3];
    const float* b2   = (const float*)d_in[4];
    const float* w3   = (const float*)d_in[5];
    const float* b3   = (const float*)d_in[6];
    const float* w4   = (const float*)d_in[7];
    const float* b4   = (const float*)d_in[8];
    const float* w5   = (const float*)d_in[9];
    const float* b5   = (const float*)d_in[10];
    const float* fc_w = (const float*)d_in[11];
    const float* fc_b = (const float*)d_in[12];
    float* out = (float*)d_out;

    // workspace carve-up (all 16B-aligned)
    _Float16* actA = (_Float16*)d_ws;                       // 71,368,704 B
    _Float16* actB = actA + ACT_HALVES;                     // 71,368,704 B
    _Float16* wf   = actB + ACT_HALVES;                     // 209,920 B (5 layers)
    float*    feat = (float*)(wf + 5 * WL_HALVES);          // 663,552 B

    // 1) zero both activation buffers (establishes the zero halo)
    {
        int n16 = (int)(2 * (size_t)ACT_HALVES * sizeof(_Float16) / 16);
        zero_ws_kernel<<<4096, 256, 0, stream>>>((float4*)actA, n16);
    }
    // 2) transpose/convert weights to f16 [layer][tap][co][ci] (tap 82 + layer1 ci>0 zero-padded)
    {
        int total = 5 * WL_HALVES;
        prep_weights_kernel<<<(total + 255) / 256, 256, 0, stream>>>(w1, w2, w3, w4, w5, wf);
    }
    // 3) center-subtract into channel 0 of padded actA
    {
        int total = NB * 100 * 81;
        prep_input_kernel<<<(total + 255) / 256, 256, 0, stream>>>(x, actA);
    }
    // 4) five conv+bias+relu layers (ping-pong A->B->A->B->A->B)
    {
        dim3 grid((T32 + 7) / 8, NB);
        conv4d_wmma_kernel<<<grid, 256, 0, stream>>>(actA, actB, wf + 0 * WL_HALVES, b1);
        conv4d_wmma_kernel<<<grid, 256, 0, stream>>>(actB, actA, wf + 1 * WL_HALVES, b2);
        conv4d_wmma_kernel<<<grid, 256, 0, stream>>>(actA, actB, wf + 2 * WL_HALVES, b3);
        conv4d_wmma_kernel<<<grid, 256, 0, stream>>>(actB, actA, wf + 3 * WL_HALVES, b4);
        conv4d_wmma_kernel<<<grid, 256, 0, stream>>>(actA, actB, wf + 4 * WL_HALVES, b5);
    }
    // 5) mean over (e,f)
    {
        int total = NB * 81 * CH;
        mean_ef_kernel<<<(total + 255) / 256, 256, 0, stream>>>(actB, feat);
    }
    // 6) FC head
    {
        int total = NB * 10 * 81;
        fc_kernel<<<(total + 255) / 256, 256, 0, stream>>>(feat, fc_w, fc_b, out);
    }
}
